// StudentAttn_offline_15410342658111
// MI455X (gfx1250) — compile-verified, matrix operand order checked by reading, not run on hardware
//
#include <hip/hip_runtime.h>

// ---------------- problem dims ----------------
#define B_   8
#define T_   64
#define S_   128
#define H_   768
#define L_   2
#define NH_  12
#define DH_  64
#define FF_  3072
#define NSEQ  (B_*T_)        // 512 sequences
#define MROWS (NSEQ*S_)      // 65536 token rows
#define QKV3  (3*H_)         // 2304

// ---------------- CDNA5 async-to-LDS (guarded; falls back to sync staging) ----
#if defined(__gfx1250__) && __has_builtin(__builtin_amdgcn_global_load_async_to_lds_b128) && __has_builtin(__builtin_amdgcn_s_wait_asynccnt)
#define USE_ASYNC 1
#else
#define USE_ASYNC 0
#endif

#if USE_ASYNC
typedef int v4i_vs __attribute__((vector_size(16)));   // matches builtin param type
typedef __attribute__((address_space(1))) v4i_vs* gptr_b128;
typedef __attribute__((address_space(3))) v4i_vs* lptr_b128;

__device__ __forceinline__ void async_cp_b128(const void* g, void* l) {
  // flat global addresses are identity; low 32 bits of a flat shared pointer
  // are the LDS byte offset (LDS aperture truncation rule).
  __builtin_amdgcn_global_load_async_to_lds_b128(
      (gptr_b128)(unsigned long long)g,
      (lptr_b128)(unsigned int)(unsigned long long)l,
      0, 0);
}
#endif

// ---------------- WMMA plumbing ----------------
typedef __bf16 bf16x16 __attribute__((ext_vector_type(16)));
typedef float  f32x8   __attribute__((ext_vector_type(8)));

union Frag { unsigned int u[8]; bf16x16 v; };

__device__ __forceinline__ unsigned short f2bf(float f) {
  unsigned u = __float_as_uint(f);
  u += 0x7fffu + ((u >> 16) & 1u);     // round-to-nearest-even
  return (unsigned short)(u >> 16);
}

// 16x32 bf16 operand fragment (A layout; B uses the same layout on a
// [col][k] tile).  Lanes 0-15: row=lane, K in {0..7,16..23}; lanes 16-31:
// row=lane-16, K in {8..15,24..31}; each VGPR holds two consecutive K.
__device__ __forceinline__ void load_frag(Frag& f, const unsigned short* base, int stride) {
  const int lane = threadIdx.x & 31;
  const unsigned short* p = base + (size_t)(lane & 15) * stride + (lane >> 4) * 8;
#pragma unroll
  for (int i = 0; i < 4; ++i) {
    f.u[i]     = *(const unsigned int*)(p + 2 * i);
    f.u[i + 4] = *(const unsigned int*)(p + 16 + 2 * i);
  }
}

__device__ __forceinline__ f32x8 wmma_bf16(const Frag& a, const Frag& b, f32x8 c) {
  return __builtin_amdgcn_wmma_f32_16x16x32_bf16(false, a.v, false, b.v, (short)0, c, false, false);
}

__device__ __forceinline__ float gelu_f(float x) {
  return 0.5f * x * (1.0f + tanhf(0.7978845608f * (x + 0.044715f * x * x * x)));
}

// -------- fp32 [K][N] -> bf16 [N][K] transposed weight conversion --------
__global__ __launch_bounds__(256) void conv_transpose_bf16_kernel(
    const float* __restrict__ src, unsigned short* __restrict__ dst, int K, int N) {
  int total = K * N;
  for (int i = blockIdx.x * 256 + threadIdx.x; i < total; i += gridDim.x * 256) {
    int k = i / N, n = i - k * N;
    dst[(size_t)n * K + k] = f2bf(src[i]);
  }
}

// ---------------- embedding + LayerNorm ----------------
__global__ __launch_bounds__(256) void embed_ln_kernel(
    const int* __restrict__ ids, const float* __restrict__ Wemb,
    const float* __restrict__ Wpos, const float* __restrict__ g,
    const float* __restrict__ b, float* __restrict__ xf,
    unsigned short* __restrict__ xh) {
  const int r = blockIdx.x;          // token row 0..65535
  const int spos = r & (S_ - 1);
  const int t = threadIdx.x;
  const int id = ids[r];
  __shared__ float rs[256], rs2[256];
  float v[3], s = 0.f, s2 = 0.f;
#pragma unroll
  for (int j = 0; j < 3; ++j) {
    int c = t + j * 256;
    float x = Wemb[(size_t)id * H_ + c] + Wpos[(size_t)spos * H_ + c];
    v[j] = x; s += x; s2 += x * x;
  }
  rs[t] = s; rs2[t] = s2; __syncthreads();
  for (int off = 128; off > 0; off >>= 1) {
    if (t < off) { rs[t] += rs[t + off]; rs2[t] += rs2[t + off]; }
    __syncthreads();
  }
  const float mu  = rs[0] * (1.0f / H_);
  const float var = rs2[0] * (1.0f / H_) - mu * mu;
  const float inv = rsqrtf(var + 1e-12f);
#pragma unroll
  for (int j = 0; j < 3; ++j) {
    int c = t + j * 256;
    float y = g[c] * (v[j] - mu) * inv + b[c];
    xf[(size_t)r * H_ + c] = y;
    xh[(size_t)r * H_ + c] = f2bf(y);
  }
}

// ---------------- residual + LayerNorm ----------------
__global__ __launch_bounds__(256) void resid_ln_kernel(
    float* __restrict__ xf, const float* __restrict__ cbuf,
    const float* __restrict__ g, const float* __restrict__ b,
    unsigned short* __restrict__ xh) {
  const int r = blockIdx.x;
  const int t = threadIdx.x;
  __shared__ float rs[256], rs2[256];
  float v[3], s = 0.f, s2 = 0.f;
#pragma unroll
  for (int j = 0; j < 3; ++j) {
    int c = t + j * 256;
    float x = xf[(size_t)r * H_ + c] + cbuf[(size_t)r * H_ + c];
    v[j] = x; s += x; s2 += x * x;
  }
  rs[t] = s; rs2[t] = s2; __syncthreads();
  for (int off = 128; off > 0; off >>= 1) {
    if (t < off) { rs[t] += rs[t + off]; rs2[t] += rs2[t + off]; }
    __syncthreads();
  }
  const float mu  = rs[0] * (1.0f / H_);
  const float var = rs2[0] * (1.0f / H_) - mu * mu;
  const float inv = rsqrtf(var + 1e-12f);
#pragma unroll
  for (int j = 0; j < 3; ++j) {
    int c = t + j * 256;
    float y = g[c] * (v[j] - mu) * inv + b[c];
    xf[(size_t)r * H_ + c] = y;
    xh[(size_t)r * H_ + c] = f2bf(y);
  }
}

// -------- tiled WMMA GEMM: C[M,N] = A[M,K] * Wt[N,K]^T (+epilogue) --------
// Double-buffered LDS; async-to-LDS staging when available.
#define EPI_F32  0
#define EPI_BF16 1
#define EPI_GELU 2

template <int EPI>
__global__ __launch_bounds__(256) void gemm_bf16_kernel(
    const unsigned short* __restrict__ A,   // [M][K] bf16 row-major
    const unsigned short* __restrict__ Wt,  // [N][K] bf16 (pre-transposed)
    const float* __restrict__ bias,         // [N]
    void* __restrict__ C,                   // [M][N] f32 or bf16
    int K, int N) {
  __shared__ unsigned short As[2][128][32];   // 2 x 8 KB
  __shared__ unsigned short Bs[2][128][32];   // 2 x 8 KB  (Bs[col][k])
  const int t = threadIdx.x;
  const int wave = t >> 5, lane = t & 31;
  const int wm = (wave >> 1) * 32;            // 4 waves along M
  const int wn = (wave & 1) * 64;             // 2 waves along N
  const int row0 = blockIdx.y * 128, col0 = blockIdx.x * 128;

  f32x8 acc[2][4];
#pragma unroll
  for (int mi = 0; mi < 2; ++mi)
#pragma unroll
    for (int ni = 0; ni < 4; ++ni)
#pragma unroll
      for (int e = 0; e < 8; ++e) acc[mi][ni][e] = 0.0f;

  auto stage = [&](int buf, int k0) {
#if USE_ASYNC
#pragma unroll
    for (int i = 0; i < 2; ++i) {            // 512 slots x 16B per tile
      int idx = t + i * 256;
      int r = idx >> 1, e8 = (idx & 1) * 8;
      async_cp_b128(&A [(size_t)(row0 + r) * K + k0 + e8], &As[buf][r][e8]);
      async_cp_b128(&Wt[(size_t)(col0 + r) * K + k0 + e8], &Bs[buf][r][e8]);
    }
#else
#pragma unroll
    for (int i = 0; i < 8; ++i) {
      int idx = t + i * 256;
      int r = idx >> 4, kp = (idx & 15) * 2;
      *(unsigned int*)&As[buf][r][kp] =
          *(const unsigned int*)&A[(size_t)(row0 + r) * K + k0 + kp];
      *(unsigned int*)&Bs[buf][r][kp] =
          *(const unsigned int*)&Wt[(size_t)(col0 + r) * K + k0 + kp];
    }
#endif
  };

  const int ntiles = K >> 5;
  stage(0, 0);
  for (int ti = 0; ti < ntiles; ++ti) {
    const int buf = ti & 1;
    const bool more = (ti + 1) < ntiles;
    if (more) stage(buf ^ 1, (ti + 1) << 5);
#if USE_ASYNC
    // per-wave in-order async completion: <=4 outstanding means the 4 copies
    // of the *current* tile have landed; barrier makes it block-wide.
    if (more) __builtin_amdgcn_s_wait_asynccnt(4);
    else      __builtin_amdgcn_s_wait_asynccnt(0);
#endif
    __syncthreads();
    Frag fa[2], fb[4];
#pragma unroll
    for (int mi = 0; mi < 2; ++mi) load_frag(fa[mi], &As[buf][wm + 16 * mi][0], 32);
#pragma unroll
    for (int ni = 0; ni < 4; ++ni) load_frag(fb[ni], &Bs[buf][wn + 16 * ni][0], 32);
#pragma unroll
    for (int mi = 0; mi < 2; ++mi)
#pragma unroll
      for (int ni = 0; ni < 4; ++ni)
        acc[mi][ni] = wmma_bf16(fa[mi], fb[ni], acc[mi][ni]);
    __syncthreads();
  }

  const int cl = lane & 15, hi = lane >> 4;
#pragma unroll
  for (int mi = 0; mi < 2; ++mi)
#pragma unroll
    for (int ni = 0; ni < 4; ++ni)
#pragma unroll
      for (int i = 0; i < 8; ++i) {
        int r = row0 + wm + 16 * mi + i + 8 * hi;
        int c = col0 + wn + 16 * ni + cl;
        float v = acc[mi][ni][i] + bias[c];
        if (EPI == EPI_GELU) v = gelu_f(v);
        if (EPI == EPI_F32)
          ((float*)C)[(size_t)r * N + c] = v;
        else
          ((unsigned short*)C)[(size_t)r * N + c] = f2bf(v);
      }
}

// ---------------- fused attention: one block per (sequence, head) ----------------
__global__ __launch_bounds__(256) void attn_kernel(
    const unsigned short* __restrict__ qkv,  // [MROWS][3H] bf16 (q|k|v)
    const int* __restrict__ mask,            // [NSEQ][S]
    unsigned short* __restrict__ O) {        // [MROWS][H] bf16 (per-head slices)
  __shared__ unsigned short Kl[S_][DH_];       // 16 KB : K rows
  __shared__ unsigned short Vt[DH_][S_];       // 16 KB : V transposed
  __shared__ unsigned short Pb[8][16][S_];     // 32 KB : per-wave softmax probs

  const int blk = blockIdx.x;
  const int n = blk / NH_, h = blk % NH_;
  const int t = threadIdx.x, wave = t >> 5, lane = t & 31;
  const int cl = lane & 15, hi = lane >> 4;
  const size_t base = (size_t)n * S_ * QKV3;

#if USE_ASYNC
#pragma unroll
  for (int i = 0; i < 4; ++i) {              // K tile: 1024 slots x 16B
    int idx = t + i * 256;
    int row = idx >> 3, d8 = (idx & 7) * 8;
    async_cp_b128(&qkv[base + (size_t)row * QKV3 + H_ + h * DH_ + d8], &Kl[row][d8]);
  }
#endif
#pragma unroll
  for (int i = 0; i < 16; ++i) {
    int idx = t + i * 256;                   // 0..4095 dword slots
    int row = idx >> 5, dp = (idx & 31) * 2;
#if !USE_ASYNC
    unsigned int kw = *(const unsigned int*)&qkv[base + (size_t)row * QKV3 + H_ + h * DH_ + dp];
    *(unsigned int*)&Kl[row][dp] = kw;
#endif
    unsigned int vw = *(const unsigned int*)&qkv[base + (size_t)row * QKV3 + 2 * H_ + h * DH_ + dp];
    Vt[dp][row]     = (unsigned short)(vw & 0xffffu);
    Vt[dp + 1][row] = (unsigned short)(vw >> 16);
  }
#if USE_ASYNC
  __builtin_amdgcn_s_wait_asynccnt(0);
#endif
  __syncthreads();

  // Q fragments for this wave's 16 query rows, straight from global
  const int mrow = wave * 16;
  const unsigned short* qrow = qkv + base + (size_t)mrow * QKV3 + h * DH_;
  Frag fq[2];
  load_frag(fq[0], qrow, QKV3);
  load_frag(fq[1], qrow + 32, QKV3);

  // scores: 16x128 held in 8 accumulators
  f32x8 sc[8];
#pragma unroll
  for (int nj = 0; nj < 8; ++nj)
#pragma unroll
    for (int e = 0; e < 8; ++e) sc[nj][e] = 0.0f;
#pragma unroll
  for (int kc = 0; kc < 2; ++kc)
#pragma unroll
    for (int nj = 0; nj < 8; ++nj) {
      Frag fk;
      load_frag(fk, &Kl[nj * 16][kc * 32], DH_);
      sc[nj] = wmma_bf16(fq[kc], fk, sc[nj]);
    }

  // key-position mask bias for this lane's 8 columns
  float bcol[8];
#pragma unroll
  for (int nj = 0; nj < 8; ++nj) {
    int col = nj * 16 + cl;
    bcol[nj] = (1.0f - (float)mask[n * S_ + col]) * -1e9f;
  }

  // row-wise softmax: reduce across the 16 lanes in this half-wave
  const float scale = 0.125f;                // 1/sqrt(64)
#pragma unroll
  for (int i = 0; i < 8; ++i) {
    float mx = -3.0e38f;
#pragma unroll
    for (int nj = 0; nj < 8; ++nj) {
      float v = sc[nj][i] * scale + bcol[nj];
      sc[nj][i] = v;
      mx = fmaxf(mx, v);
    }
    for (int off = 1; off < 16; off <<= 1) mx = fmaxf(mx, __shfl_xor(mx, off));
    float ssum = 0.f;
#pragma unroll
    for (int nj = 0; nj < 8; ++nj) {
      float e = __expf(sc[nj][i] - mx);
      sc[nj][i] = e;
      ssum += e;
    }
    for (int off = 1; off < 16; off <<= 1) ssum += __shfl_xor(ssum, off);
    float inv = 1.0f / ssum;
#pragma unroll
    for (int nj = 0; nj < 8; ++nj) sc[nj][i] *= inv;
  }

  // repack probs into this wave's LDS A-tile (wave-private, DS ops in-order)
#pragma unroll
  for (int nj = 0; nj < 8; ++nj)
#pragma unroll
    for (int i = 0; i < 8; ++i)
      Pb[wave][i + 8 * hi][nj * 16 + cl] = f2bf(sc[nj][i]);

  // O = P * V : 16x64, K-dim = 128 keys
  f32x8 oc[4];
#pragma unroll
  for (int nd = 0; nd < 4; ++nd)
#pragma unroll
    for (int e = 0; e < 8; ++e) oc[nd][e] = 0.0f;
#pragma unroll
  for (int kc = 0; kc < 4; ++kc) {
    Frag fp;
    load_frag(fp, &Pb[wave][0][kc * 32], S_);
#pragma unroll
    for (int nd = 0; nd < 4; ++nd) {
      Frag fv;
      load_frag(fv, &Vt[nd * 16][kc * 32], S_);
      oc[nd] = wmma_bf16(fp, fv, oc[nd]);
    }
  }
#pragma unroll
  for (int nd = 0; nd < 4; ++nd)
#pragma unroll
    for (int i = 0; i < 8; ++i) {
      int row = n * S_ + mrow + i + 8 * hi;
      int col = h * DH_ + nd * 16 + cl;
      O[(size_t)row * H_ + col] = f2bf(oc[nd][i]);
    }
}

// ---------------- masked mean pooling over tokens ----------------
__global__ __launch_bounds__(256) void pool_kernel(const float* __restrict__ xf,
                                                   const int* __restrict__ mask,
                                                   float* __restrict__ pooled) {
  const int n = blockIdx.x;
  const int t = threadIdx.x;
  float acc[3] = {0.f, 0.f, 0.f};
  float msum = 0.f;
  for (int s = 0; s < S_; ++s) {
    float m = (float)mask[n * S_ + s];
    msum += m;
    const float* row = xf + ((size_t)n * S_ + s) * H_;
#pragma unroll
    for (int j = 0; j < 3; ++j) acc[j] += row[t + j * 256] * m;
  }
  float inv = 1.0f / fmaxf(msum, 1.0f);
#pragma unroll
  for (int j = 0; j < 3; ++j)
    pooled[(size_t)n * H_ + t + j * 256] = acc[j] * inv;
}

// ---------------- per-user head: tweet attention + classifier + outputs ----------------
#define OUT_LOGITS 0
#define OUT_EMB_S  16
#define OUT_EMB_T  (16 + B_ * T_ * H_)
#define OUT_SCORES (16 + 2 * B_ * T_ * H_)

__global__ __launch_bounds__(256) void head_kernel(
    const float* __restrict__ pooled, const int* __restrict__ len_real,
    const float* __restrict__ teach, const float* __restrict__ w_attn,
    const float* __restrict__ b_attn, const float* __restrict__ W_cls,
    const float* __restrict__ b_cls, float* __restrict__ out) {
  const int b = blockIdx.x;
  const int t = threadIdx.x;
  __shared__ float sc[T_], scs[T_], eu[H_];
  int lr = len_real[b]; if (lr < 1) lr = 1;

  if (t < T_) {
    const float* p = pooled + (size_t)(b * T_ + t) * H_;
    float s = 0.f;
    for (int j = 0; j < H_; ++j) s += p[j] * w_attn[j];
    s += b_attn[0];
    sc[t] = (t < lr) ? s : -1e9f;
  }
  __syncthreads();

  float mx = -3.0e38f;
  for (int i = 0; i < T_; ++i) mx = fmaxf(mx, sc[i]);
  float ssum = 0.f;
  for (int i = 0; i < T_; ++i) ssum += __expf(sc[i] - mx);
  if (t < T_) {
    float v = (t < lr) ? __expf(sc[t] - mx) / ssum : 0.0f;
    scs[t] = v;
    out[OUT_SCORES + b * T_ + t] = v;
  }
  __syncthreads();

#pragma unroll
  for (int j = 0; j < 3; ++j) {
    int c = t + j * 256;
    float a = 0.f;
    for (int i = 0; i < T_; ++i) a += scs[i] * pooled[(size_t)(b * T_ + i) * H_ + c];
    eu[c] = a;
  }

  for (int e = t; e < T_ * H_; e += 256) {
    int tt = e / H_, c = e - tt * H_;
    bool val = tt < lr;
    size_t src = (size_t)(b * T_ + tt) * H_ + c;
    out[OUT_EMB_S + src] = val ? pooled[src] : 0.0f;
    out[OUT_EMB_T + src] = val ? teach[src] : 0.0f;
  }
  __syncthreads();

  if (t < 2) {
    float a = 0.f;
    for (int j = 0; j < H_; ++j) a += eu[j] * W_cls[j * 2 + t];
    out[OUT_LOGITS + b * 2 + t] = a + b_cls[t];
  }
}

// ---------------- host-side orchestration ----------------
extern "C" void kernel_launch(void* const* d_in, const int* in_sizes, int n_in,
                              void* d_out, int out_size, void* d_ws, size_t ws_size,
                              hipStream_t stream) {
  (void)in_sizes; (void)n_in; (void)out_size; (void)ws_size;
  const int*   ids    = (const int*)d_in[0];
  const int*   amask  = (const int*)d_in[1];
  const int*   lenr   = (const int*)d_in[2];
  const float* teach  = (const float*)d_in[3];
  const float* Wemb   = (const float*)d_in[4];
  const float* Wpos   = (const float*)d_in[5];
  const float* g_emb  = (const float*)d_in[6];
  const float* b_emb  = (const float*)d_in[7];
  const float* Wqkv   = (const float*)d_in[8];
  const float* bqkv   = (const float*)d_in[9];
  const float* Wo     = (const float*)d_in[10];
  const float* bo     = (const float*)d_in[11];
  const float* g1     = (const float*)d_in[12];
  const float* b1     = (const float*)d_in[13];
  const float* W1     = (const float*)d_in[14];
  const float* bf1    = (const float*)d_in[15];
  const float* W2     = (const float*)d_in[16];
  const float* bf2    = (const float*)d_in[17];
  const float* g2     = (const float*)d_in[18];
  const float* b2     = (const float*)d_in[19];
  const float* w_attn = (const float*)d_in[20];
  const float* b_attn = (const float*)d_in[21];
  const float* W_cls  = (const float*)d_in[22];
  const float* b_cls  = (const float*)d_in[23];
  float* out = (float*)d_out;

  char* ws = (char*)d_ws;
  size_t off = 0;
  auto carve = [&](size_t bytes) -> char* {
    char* p = ws + off;
    off += (bytes + 255) & ~(size_t)255;
    return p;
  };
  float*          x_f    = (float*)carve((size_t)MROWS * H_ * 4);
  unsigned short* x_h    = (unsigned short*)carve((size_t)MROWS * H_ * 2);
  unsigned short* qkv_h  = (unsigned short*)carve((size_t)MROWS * QKV3 * 2);
  unsigned short* o_h    = (unsigned short*)carve((size_t)MROWS * H_ * 2);
  float*          c_f    = (float*)carve((size_t)MROWS * H_ * 4);
  unsigned short* h_h    = (unsigned short*)carve((size_t)MROWS * FF_ * 2);
  float*          pooled = (float*)carve((size_t)NSEQ * H_ * 4);
  unsigned short* wqkv_h = (unsigned short*)carve((size_t)L_ * H_ * QKV3 * 2);
  unsigned short* wo_h   = (unsigned short*)carve((size_t)L_ * H_ * H_ * 2);
  unsigned short* w1_h   = (unsigned short*)carve((size_t)L_ * H_ * FF_ * 2);
  unsigned short* w2_h   = (unsigned short*)carve((size_t)L_ * FF_ * H_ * 2);

  auto convT = [&](const float* s, unsigned short* d, int Kd, int Nd) {
    int n = Kd * Nd;
    int blocks = (n + 255) / 256;
    if (blocks > 16384) blocks = 16384;
    conv_transpose_bf16_kernel<<<blocks, 256, 0, stream>>>(s, d, Kd, Nd);
  };
  for (int l = 0; l < L_; ++l) {
    convT(Wqkv + (size_t)l * H_ * QKV3, wqkv_h + (size_t)l * H_ * QKV3, H_, QKV3);
    convT(Wo   + (size_t)l * H_ * H_,   wo_h   + (size_t)l * H_ * H_,   H_, H_);
    convT(W1   + (size_t)l * H_ * FF_,  w1_h   + (size_t)l * H_ * FF_,  H_, FF_);
    convT(W2   + (size_t)l * FF_ * H_,  w2_h   + (size_t)l * FF_ * H_,  FF_, H_);
  }

  embed_ln_kernel<<<MROWS, 256, 0, stream>>>(ids, Wemb, Wpos, g_emb, b_emb, x_f, x_h);

  for (int l = 0; l < L_; ++l) {
    gemm_bf16_kernel<EPI_BF16><<<dim3(QKV3 / 128, MROWS / 128), 256, 0, stream>>>(
        x_h, wqkv_h + (size_t)l * H_ * QKV3, bqkv + (size_t)l * QKV3, qkv_h, H_, QKV3);
    attn_kernel<<<NSEQ * NH_, 256, 0, stream>>>(qkv_h, amask, o_h);
    gemm_bf16_kernel<EPI_F32><<<dim3(H_ / 128, MROWS / 128), 256, 0, stream>>>(
        o_h, wo_h + (size_t)l * H_ * H_, bo + (size_t)l * H_, c_f, H_, H_);
    resid_ln_kernel<<<MROWS, 256, 0, stream>>>(x_f, c_f, g1 + (size_t)l * H_, b1 + (size_t)l * H_, x_h);
    gemm_bf16_kernel<EPI_GELU><<<dim3(FF_ / 128, MROWS / 128), 256, 0, stream>>>(
        x_h, w1_h + (size_t)l * H_ * FF_, bf1 + (size_t)l * FF_, h_h, H_, FF_);
    gemm_bf16_kernel<EPI_F32><<<dim3(H_ / 128, MROWS / 128), 256, 0, stream>>>(
        h_h, w2_h + (size_t)l * FF_ * H_, bf2 + (size_t)l * H_, c_f, FF_, H_);
    resid_ln_kernel<<<MROWS, 256, 0, stream>>>(x_f, c_f, g2 + (size_t)l * H_, b2 + (size_t)l * H_, x_h);
  }

  pool_kernel<<<NSEQ, 256, 0, stream>>>(x_f, amask, pooled);
  head_kernel<<<B_, 256, 0, stream>>>(pooled, lenr, teach, w_attn, b_attn, W_cls, b_cls, out);
}
